// NTXentLoss_12824772346324
// MI455X (gfx1250) — matrix operand order repaired and checked by reading, not compiled, not run on hardware
//
#include <hip/hip_runtime.h>
#include <hip/hip_bf16.h>

typedef __attribute__((ext_vector_type(16))) _Float16 v16h;
typedef __attribute__((ext_vector_type(8)))  _Float16 v8h;
typedef __attribute__((ext_vector_type(8)))  float    v8f;
typedef __attribute__((ext_vector_type(4)))  int      v4i;

#define NROWS   8192
#define BATCH   4096
#define DIM     512
#define INVT    2.0f          // 1 / TEMPERATURE
#define SHIFT   2.0f          // logits <= 1/T for unit vectors -> fixed softmax shift
#define NSPLIT  8
#define NRANGE  (NROWS / NSPLIT)   // 1024 columns per block
#define BN      64                 // columns per staged tile (ping-pong buffered)
#define LDSTRIDE 520               // padded halves per staged column (kills LDS bank conflicts)
#define BUFHALVES (BN * LDSTRIDE)  // 33280 halves = 66560 B per buffer

// ---- CDNA5 async global->LDS path (guarded; falls back to manual copy) ----
#if defined(__gfx1250__) && __has_builtin(__builtin_amdgcn_global_load_async_to_lds_b128)
#define HAVE_ASYNC_LDS 1
typedef __attribute__((address_space(1))) v4i GV4;   // global 128-bit chunk
typedef __attribute__((address_space(3))) v4i LV4;   // LDS 128-bit chunk
#else
#define HAVE_ASYNC_LDS 0
#endif

#if defined(__gfx1250__) && __has_builtin(__builtin_amdgcn_s_wait_asynccnt)
#define WAIT_ASYNC(n) __builtin_amdgcn_s_wait_asynccnt(n)
#elif defined(__gfx1250__)
#define WAIT_ASYNC(n) asm volatile("s_wait_asynccnt %0" ::"i"(n) : "memory")
#else
#define WAIT_ASYNC(n)
#endif

// ---------------- Kernel 1: L2-normalize rows, emit f16 ----------------
__global__ void ntx_normalize(const float* __restrict__ z1,
                              const float* __restrict__ z2,
                              _Float16* __restrict__ zn) {
  const int lane = threadIdx.x & 31;
  const int row  = (int)((blockIdx.x * blockDim.x + threadIdx.x) >> 5);
  const float* src = (row < BATCH) ? (z1 + (size_t)row * DIM)
                                   : (z2 + (size_t)(row - BATCH) * DIM);
  const float* p = src + lane * 16;   // 16 contiguous floats per lane (512/32)
  float vals[16];
  float ss = 0.0f;
#pragma unroll
  for (int j = 0; j < 16; ++j) { vals[j] = p[j]; ss += vals[j] * vals[j]; }
#pragma unroll
  for (int off = 16; off > 0; off >>= 1) ss += __shfl_xor(ss, off, 32);
  const float inv = rsqrtf(fmaxf(ss, 1e-24f));   // matches max(||z||, 1e-12)
  v8h o0, o1;
#pragma unroll
  for (int j = 0; j < 8; ++j) {
    o0[j] = (_Float16)(vals[j] * inv);
    o1[j] = (_Float16)(vals[j + 8] * inv);
  }
  _Float16* dst = zn + (size_t)row * DIM + lane * 16;
  *(v8h*)(dst)     = o0;
  *(v8h*)(dst + 8) = o1;
}

// Stage BN columns (zn rows) into an LDS buffer with padded stride.
// Async path: 16 b128 async-LDS loads per thread (tracked by ASYNCcnt).
__device__ __forceinline__ void stage_tile(const _Float16* __restrict__ gsrc,
                                           _Float16* __restrict__ lbuf,
                                           int tid) {
#pragma unroll
  for (int j = 0; j < 16; ++j) {
    const int c   = tid + 256 * j;       // 16-byte chunk id, 4096 total
    const int col = c >> 6;              // 64 chunks per column
    const int seg = c & 63;
    const _Float16* g = gsrc + (size_t)col * DIM + seg * 8;
    _Float16*       l = lbuf + col * LDSTRIDE + seg * 8;
#if HAVE_ASYNC_LDS
    GV4* gp = (GV4*)(__UINTPTR_TYPE__)(const void*)g;
    LV4* lp = (LV4*)(unsigned int)(__UINTPTR_TYPE__)(void*)l;
    __builtin_amdgcn_global_load_async_to_lds_b128(gp, lp, 0, 0);
#else
    *(v8h*)l = *(const v8h*)g;
#endif
  }
}

// ---- Kernel 2: fused  z @ z^T / T  +  shifted exp-sum (per row, per split) ----
// Block: 256 threads = 8 waves; wave w owns rows [bx*128 + 16w, +16).
// Grid.y splits the 8192 columns into NSPLIT independent ranges (sums add).
__global__ __launch_bounds__(256)
void ntx_simlse(const _Float16* __restrict__ zn, float* __restrict__ spart) {
  extern __shared__ _Float16 bsm[];            // 2 * BUFHALVES halves (~130 KB)
  const int tid  = threadIdx.x;
  const int lane = tid & 31;
  const int wave = tid >> 5;
  const int R    = (int)blockIdx.x * 128 + wave * 16;
  const int nbeg = (int)blockIdx.y * NRANGE;

  // A fragments (16 rows x full K=512) resident in VGPRs for the whole kernel.
  // ISA layout: lane l holds row (l&15); halves 0..7 = K chunk + (l>>4)*8,
  // halves 8..15 = K chunk + 16 + (l>>4)*8.
  v16h afrag[16];
  {
    const _Float16* arow = zn + (size_t)(R + (lane & 15)) * DIM;
    const int ks = (lane >> 4) * 8;
#pragma unroll
    for (int c = 0; c < 16; ++c) {
      v8h lo = *(const v8h*)(arow + c * 32 + ks);
      v8h hi = *(const v8h*)(arow + c * 32 + 16 + ks);
      afrag[c] = __builtin_shufflevector(lo, hi,
                   0, 1, 2, 3, 4, 5, 6, 7, 8, 9, 10, 11, 12, 13, 14, 15);
    }
  }

  float srun[8];
#pragma unroll
  for (int r = 0; r < 8; ++r) srun[r] = 0.0f;

  const int gm = R + ((lane >> 4) << 3);       // global row index of c[0] for this lane
  const int ntiles = NRANGE / BN;              // 16

  // Prime the ping-pong pipeline with tile 0.
  stage_tile(zn + (size_t)nbeg * DIM, bsm, tid);

  for (int t = 0; t < ntiles; ++t) {
    const int n0 = nbeg + t * BN;
    _Float16* cur = bsm + (size_t)(t & 1) * BUFHALVES;

    if (t + 1 < ntiles) {
      // Kick off the next tile's async prefetch into the other buffer,
      // then wait until only those 16 newest ops remain in flight
      // (async loads complete in order -> current tile is resident).
      stage_tile(zn + (size_t)(n0 + BN) * DIM,
                 bsm + (size_t)((t + 1) & 1) * BUFHALVES, tid);
      WAIT_ASYNC(16);
    } else {
      WAIT_ASYNC(0);
    }
    __syncthreads();   // all waves' portions of the current tile are visible

#pragma unroll
    for (int nt = 0; nt < BN / 16; ++nt) {
      // B fragment: lane holds column (nt*16 + (l&15)), K at (l>>4)*8 and 16 + (l>>4)*8.
      const _Float16* bcol =
          cur + (nt * 16 + (lane & 15)) * LDSTRIDE + (lane >> 4) * 8;
      v8f c = {};
#pragma unroll
      for (int kc = 0; kc < 16; ++kc) {
        v8h blo = *(const v8h*)(bcol + kc * 32);
        v8h bhi = *(const v8h*)(bcol + kc * 32 + 16);
        v16h b = __builtin_shufflevector(blo, bhi,
                   0, 1, 2, 3, 4, 5, 6, 7, 8, 9, 10, 11, 12, 13, 14, 15);
        c = __builtin_amdgcn_wmma_f32_16x16x32_f16(
                false, afrag[kc], false, b, (short)0, c, false, false);
      }
      const int gn = n0 + nt * 16 + (lane & 15);
#pragma unroll
      for (int r = 0; r < 8; ++r) {
        float e = __expf(fmaf(c[r], INVT, -SHIFT));   // exp(sim - 2), in (0, 1]
        e = (gm + r == gn) ? 0.0f : e;                // mask diagonal
        srun[r] += e;
      }
    }
    __syncthreads();   // done reading `cur`; safe to overwrite next iteration
  }

  // Row sums: combine the 16 lanes sharing each output row.
#pragma unroll
  for (int r = 0; r < 8; ++r) {
    float s = srun[r];
#pragma unroll
    for (int off = 1; off < 16; off <<= 1) s += __shfl_xor(s, off, 16);
    if ((lane & 15) == 0)
      spart[(size_t)(gm + r) * NSPLIT + blockIdx.y] = s;
  }
}

// ---- Kernel 3: positive logit + per-row NLL, per-block partial sums ----
__global__ void ntx_nll(const _Float16* __restrict__ zn,
                        const float* __restrict__ spart,
                        float* __restrict__ partial) {
  const int lane = threadIdx.x & 31;
  const int wave = (int)threadIdx.x >> 5;
  const int row  = (int)blockIdx.x * 8 + wave;
  const int partner = row ^ BATCH;             // i <-> i +/- 4096
  const _Float16* a = zn + (size_t)row * DIM + lane * 16;
  const _Float16* b = zn + (size_t)partner * DIM + lane * 16;
  float acc = 0.0f;
#pragma unroll
  for (int j = 0; j < 16; ++j) acc += (float)a[j] * (float)b[j];
#pragma unroll
  for (int off = 16; off > 0; off >>= 1) acc += __shfl_xor(acc, off, 32);
  __shared__ float wpart[8];
  if (lane == 0) {
    float s = 0.0f;
#pragma unroll
    for (int g = 0; g < NSPLIT; ++g) s += spart[(size_t)row * NSPLIT + g];
    wpart[wave] = (SHIFT + __logf(s)) - INVT * acc;   // lse - pos
  }
  __syncthreads();
  if (threadIdx.x == 0) {
    float s = 0.0f;
#pragma unroll
    for (int w = 0; w < 8; ++w) s += wpart[w];
    partial[blockIdx.x] = s;
  }
}

// ---- Kernel 4: deterministic final mean over 1024 partials ----
__global__ void ntx_final(const float* __restrict__ partial, float* __restrict__ out) {
  __shared__ float sm[256];
  const int t = threadIdx.x;
  sm[t] = partial[t] + partial[t + 256] + partial[t + 512] + partial[t + 768];
  __syncthreads();
  for (int off = 128; off > 0; off >>= 1) {
    if (t < off) sm[t] += sm[t + off];
    __syncthreads();
  }
  if (t == 0) out[0] = sm[0] * (1.0f / (float)NROWS);
}

extern "C" void kernel_launch(void* const* d_in, const int* in_sizes, int n_in,
                              void* d_out, int out_size, void* d_ws, size_t ws_size,
                              hipStream_t stream) {
  const float* z1 = (const float*)d_in[0];
  const float* z2 = (const float*)d_in[1];

  char* ws = (char*)d_ws;
  _Float16* zn   = (_Float16*)ws;                                   // 8 MB
  float* spart   = (float*)(ws + (size_t)NROWS * DIM * 2);          // 256 KB
  float* partial = (float*)(ws + (size_t)NROWS * DIM * 2
                               + (size_t)NROWS * NSPLIT * 4);       // 4 KB

  ntx_normalize<<<NROWS / 8, 256, 0, stream>>>(z1, z2, zn);

  const size_t lds = (size_t)2 * BUFHALVES * sizeof(_Float16);      // ~130 KB < 320 KB WGP LDS
  (void)hipFuncSetAttribute(reinterpret_cast<const void*>(ntx_simlse),
                            hipFuncAttributeMaxDynamicSharedMemorySize, (int)lds);
  ntx_simlse<<<dim3(NROWS / 128, NSPLIT), 256, lds, stream>>>(zn, spart);

  ntx_nll<<<NROWS / 8, 256, 0, stream>>>(zn, spart, partial);
  ntx_final<<<1, 256, 0, stream>>>(partial, (float*)d_out);
}
